// MidigenMamba_42528766165466
// MI455X (gfx1250) — matrix operand order repaired and checked by reading, not compiled, not run on hardware
//
#include <hip/hip_runtime.h>

// Problem constants (from reference)
#define DEPTH 6
#define DIMM  768
#define EXPD  1536           // E
#define NST   16             // N
#define KCONV 4
#define RRANK 48
#define VOC   1024
#define BATCH 2
#define SEQ   2048
#define BL    (BATCH*SEQ)    // 4096 tokens
#define DBL_LD (RRANK + 2*NST)  // 80

typedef __attribute__((ext_vector_type(16))) __bf16 bf16x16;
typedef __attribute__((ext_vector_type(8)))  float  v8f;
typedef __attribute__((ext_vector_type(4)))  unsigned int u32x4;
typedef __attribute__((ext_vector_type(8)))  int          i32x8;
typedef __attribute__((ext_vector_type(4)))  int          i32x4;

#if __has_builtin(__builtin_amdgcn_tensor_load_to_lds)
#define HAS_TDM 1
#else
#define HAS_TDM 0
#endif

static __device__ __forceinline__ unsigned int pack_bf16x2(float a, float b) {
  unsigned int ua = __float_as_uint(a);
  unsigned int ub = __float_as_uint(b);
  ua = (ua + 0x7FFFu + ((ua >> 16) & 1u)) >> 16;   // RNE
  ub = (ub + 0x7FFFu + ((ub >> 16) & 1u)) >> 16;
  return (ua & 0xFFFFu) | (ub << 16);
}
static __device__ __forceinline__ unsigned short f2bf(float a) {
  unsigned int ua = __float_as_uint(a);
  return (unsigned short)((ua + 0x7FFFu + ((ua >> 16) & 1u)) >> 16);
}

#if HAS_TDM
// Issue a TDM 2D tile load: bf16 tile (64 k) x (128 n-rows) from row-major
// Bt[N][Kd] into LDS at ldsOff, with 16B LDS padding after each 128B row
// (-> 144B row stride). OOB rows/cols (K=48, N=80 cases) read as zero.
static __device__ __forceinline__ void tdm_load_b_tile(
    const unsigned short* __restrict__ Bt, int Kd, int N,
    int colBase, int k0, unsigned ldsOff)
{
  unsigned long long ga =
      (unsigned long long)(uintptr_t)(Bt + (size_t)colBase * Kd + k0);
  unsigned tdim0 = (unsigned)(Kd - k0);            // remaining k from tile start
  unsigned tdim1 = (unsigned)(N - colBase);        // remaining rows
  // D# group 0: count=1 | lds_addr | global_addr(57b) | type=2
  u32x4 g0 = { 1u, ldsOff, (unsigned)ga,
               (unsigned)((ga >> 32) & 0x01FFFFFFull) | (2u << 30) };
  // D# group 1: data_size=2B, pad_enable, pad_interval=4 (128B), pad_amount=3 (16B)
  unsigned w4 = (1u << 16) | (1u << 20) | (4u << 22) | (3u << 25);
  i32x8 g1 = { (int)w4,
               (int)((tdim0 & 0xFFFFu) << 16),                    // tensor_dim0 lo
               (int)(((tdim0 >> 16) & 0xFFFFu) | ((tdim1 & 0xFFFFu) << 16)),
               (int)(((tdim1 >> 16) & 0xFFFFu) | (64u << 16)),    // tile_dim0=64
               (int)128u,                                         // tile_dim1=128
               (int)(unsigned)Kd,                                 // dim0_stride
               0, 0 };
  i32x4 gz = { 0, 0, 0, 0 };
#if defined(__clang_major__) && (__clang_major__ >= 23)
  i32x8 z8 = { 0, 0, 0, 0, 0, 0, 0, 0 };
  __builtin_amdgcn_tensor_load_to_lds(g0, g1, gz, gz, z8, 0);
#else
  __builtin_amdgcn_tensor_load_to_lds(g0, g1, gz, gz, 0);
#endif
}
#endif

// ---------------------------------------------------------------------------
// bf16 WMMA GEMM: C[M,N] = A[M,K] * Bt[N,K]^T  (A fp32, Bt pre-converted bf16)
// EPI: 0 = store, 1 = += extra[row,col] (residual), 2 = softplus(acc+extra[col])
// KALIGNED: K % 64 == 0 (guard-free A staging; M always 4096 here).
// Block tile 128x128, BK=64, 8 waves (wave32), wave tile 32x64.
// B tile fetched by the Tensor Data Mover (TENSORcnt), A staged by VALU.
// ---------------------------------------------------------------------------
template<int EPI, bool KALIGNED>
__global__ __launch_bounds__(256, 1)
void gemm_wmma_bf16(const float* __restrict__ A, int lda,
                    const unsigned short* __restrict__ Bt,
                    float* __restrict__ C, int ldc,
                    const float* __restrict__ extra,
                    int M, int N, int Kd)
{
  // 72 = 64 + 8 pad ushorts: 144B row stride -> conflict-free b128 reads,
  // and exactly what the TDM pad_interval/pad_amount settings produce.
  __shared__ unsigned short sA[128][72];   // row-major [m][k]
  __shared__ unsigned short sB[128][72];   // transposed [n][k]

  const int tid  = threadIdx.x;
  const int lane = tid & 31;
  const int wave = tid >> 5;
  const int wm   = wave >> 1;   // 0..3 : 32-row strip
  const int wn   = wave & 1;    // 0..1 : 64-col strip
  const int l16  = lane & 15;
  const int hi   = lane >> 4;

  const int rowBase = blockIdx.y * 128;
  const int colBase = blockIdx.x * 128;
#if HAS_TDM
  const unsigned ldsB = (unsigned)(uintptr_t)(&sB[0][0]);   // low 32b = LDS offset
#endif

  v8f acc[2][4];
  {
    v8f z = {0.f,0.f,0.f,0.f,0.f,0.f,0.f,0.f};
    #pragma unroll
    for (int i = 0; i < 2; ++i)
      #pragma unroll
      for (int j = 0; j < 4; ++j) acc[i][j] = z;
  }

  const int kTiles = (Kd + 63) >> 6;
  for (int kb = 0; kb < kTiles; ++kb) {
    const int k0 = kb << 6;
    __syncthreads();   // previous iteration's LDS reads complete

#if HAS_TDM
    // ---- B tile via Tensor Data Mover (one wave issues; EXEC-independent) ----
    if (wave == 0)
      tdm_load_b_tile(Bt, Kd, N, colBase, k0, ldsB);
#else
    // ---- fallback: cooperative bf16 copy of B tile ----
    #pragma unroll
    for (int it = 0; it < 4; ++it) {
      int c = it * 256 + tid;           // 1024 8-elem chunks
      int n = c >> 3;
      int k = (c & 7) << 3;
      uint4 v = make_uint4(0u, 0u, 0u, 0u);
      int gn = colBase + n, gk = k0 + k;
      if (gn < N) {
        if (gk + 7 < Kd) {
          v = *reinterpret_cast<const uint4*>(Bt + (size_t)gn * Kd + gk);
        } else {
          unsigned short* e = reinterpret_cast<unsigned short*>(&v);
          for (int j = 0; j < 8; ++j)
            if (gk + j < Kd) e[j] = Bt[(size_t)gn * Kd + gk + j];
        }
      }
      *reinterpret_cast<uint4*>(&sB[n][k]) = v;
    }
#endif

    // ---- stage A tile 128x64, fp32 -> bf16 ----
    #pragma unroll
    for (int it = 0; it < 8; ++it) {
      int c   = it * 256 + tid;         // 2048 float4 chunks
      int row = c >> 4;                 // 16 chunks per row
      int col = (c & 15) << 2;
      float4 v;
      if (KALIGNED) {
        v = *reinterpret_cast<const float4*>(A + (size_t)(rowBase + row) * lda + k0 + col);
      } else {
        v = make_float4(0.f, 0.f, 0.f, 0.f);
        int gr = rowBase + row, gc = k0 + col;
        if (gr < M) {
          if (gc + 3 < Kd) {
            v = *reinterpret_cast<const float4*>(A + (size_t)gr * lda + gc);
          } else {
            if (gc + 0 < Kd) v.x = A[(size_t)gr * lda + gc + 0];
            if (gc + 1 < Kd) v.y = A[(size_t)gr * lda + gc + 1];
            if (gc + 2 < Kd) v.z = A[(size_t)gr * lda + gc + 2];
            if (gc + 3 < Kd) v.w = A[(size_t)gr * lda + gc + 3];
          }
        }
      }
      *reinterpret_cast<uint2*>(&sA[row][col]) =
          make_uint2(pack_bf16x2(v.x, v.y), pack_bf16x2(v.z, v.w));
    }

    // prefetch next A k-tile into cache (global_prefetch_b8)
    if (kb + 1 < kTiles)
      __builtin_prefetch(A + (size_t)(rowBase + (tid & 127)) * lda + k0 + 64, 0, 1);

#if HAS_TDM
    if (wave == 0)
      __builtin_amdgcn_s_wait_tensorcnt(0);     // TDM tile landed in LDS
#endif
    __syncthreads();

    // ---- two k-steps of 32 (fragment layouts per ISA §7.12.2) ----
    #pragma unroll
    for (int ks = 0; ks < 2; ++ks) {
      const int ko = ks << 5;
      union Frag { uint4 q[2]; bf16x16 v; } fa[2], fb[4];
      #pragma unroll
      for (int tm = 0; tm < 2; ++tm) {
        // lane m(hi=0): K {0..7,16..23}; lane m+16: K {8..15,24..31}
        const unsigned short* p = &sA[wm * 32 + tm * 16 + l16][ko + hi * 8];
        fa[tm].q[0] = *reinterpret_cast<const uint4*>(p);
        fa[tm].q[1] = *reinterpret_cast<const uint4*>(p + 16);
      }
      #pragma unroll
      for (int tn = 0; tn < 4; ++tn) {
        // lane n(hi=0): K {0..7,16..23}; lane n+16: K {8..15,24..31}
        const unsigned short* p = &sB[wn * 64 + tn * 16 + l16][ko + hi * 8];
        fb[tn].q[0] = *reinterpret_cast<const uint4*>(p);
        fb[tn].q[1] = *reinterpret_cast<const uint4*>(p + 16);
      }
      #pragma unroll
      for (int tm = 0; tm < 2; ++tm)
        #pragma unroll
        for (int tn = 0; tn < 4; ++tn)
          acc[tm][tn] = __builtin_amdgcn_wmma_f32_16x16x32_bf16(
              false, fa[tm].v, false, fb[tn].v, (short)0, acc[tm][tn],
              false, false);
    }
  }

  // ---- epilogue: C layout = lane(n=l16, hi) x VGPR r -> m = hi*8 + r ----
  #pragma unroll
  for (int tm = 0; tm < 2; ++tm) {
    #pragma unroll
    for (int tn = 0; tn < 4; ++tn) {
      int col = colBase + wn * 64 + tn * 16 + l16;
      if (col >= N) continue;
      int rbase = rowBase + wm * 32 + tm * 16 + hi * 8;
      #pragma unroll
      for (int r = 0; r < 8; ++r) {
        int row = rbase + r;
        if (row >= M) continue;
        float v = acc[tm][tn][r];
        if (EPI == 1) v += extra[(size_t)row * ldc + col];
        if (EPI == 2) {
          v += extra[col];
          v = (v > 20.f) ? v : log1pf(__expf(v));   // softplus
        }
        C[(size_t)row * ldc + col] = v;
      }
    }
  }
}

// ---------------------------------------------------------------------------
// Weight prep: out[n*K + k] = bf16(W[k*N + n])   (transpose + convert)
// ---------------------------------------------------------------------------
__global__ __launch_bounds__(256)
void transpose_bf16_kernel(const float* __restrict__ W,
                           unsigned short* __restrict__ out, int K, int N)
{
  int i = blockIdx.x * blockDim.x + threadIdx.x;
  if (i >= N * K) return;
  int n = i / K, k = i % K;
  out[i] = f2bf(W[(size_t)k * N + n]);
}

// ---------------------------------------------------------------------------
// Embedding: x[b,l,:] = token_emb[id] + pos_emb[l]
// ---------------------------------------------------------------------------
__global__ __launch_bounds__(256)
void embed_kernel(const int* __restrict__ ids, const float* __restrict__ tok,
                  const float* __restrict__ pos, float* __restrict__ x)
{
  int i = blockIdx.x * blockDim.x + threadIdx.x;
  if (i >= BL * DIMM) return;
  int d  = i % DIMM;
  int bl = i / DIMM;
  int t  = bl % SEQ;
  x[i] = tok[(size_t)ids[bl] * DIMM + d] + pos[(size_t)t * DIMM + d];
}

// ---------------------------------------------------------------------------
// LayerNorm: one 256-thread block per token (768 features)
// ---------------------------------------------------------------------------
__global__ __launch_bounds__(256)
void layernorm_kernel(const float* __restrict__ x, const float* __restrict__ g,
                      const float* __restrict__ b, float* __restrict__ y)
{
  const int tok = blockIdx.x;
  const float* xr = x + (size_t)tok * DIMM;
  float s = 0.f, ss = 0.f;
  for (int i = threadIdx.x; i < DIMM; i += 256) {
    float v = xr[i]; s += v; ss += v * v;
  }
  __shared__ float rs[8], rss[8];
  #pragma unroll
  for (int o = 16; o > 0; o >>= 1) {
    s  += __shfl_down(s,  o, 32);
    ss += __shfl_down(ss, o, 32);
  }
  int lane = threadIdx.x & 31, wave = threadIdx.x >> 5;
  if (lane == 0) { rs[wave] = s; rss[wave] = ss; }
  __syncthreads();
  s = 0.f; ss = 0.f;
  #pragma unroll
  for (int w = 0; w < 8; ++w) { s += rs[w]; ss += rss[w]; }
  float mean = s * (1.f / DIMM);
  float var  = ss * (1.f / DIMM) - mean * mean;
  float rstd = rsqrtf(var + 1e-5f);
  float* yr = y + (size_t)tok * DIMM;
  for (int i = threadIdx.x; i < DIMM; i += 256)
    yr[i] = (xr[i] - mean) * rstd * g[i] + b[i];
}

// ---------------------------------------------------------------------------
// Causal depthwise conv (K=4) + SiLU. u = xz[..., :E] with ld 2E.
// ---------------------------------------------------------------------------
__global__ __launch_bounds__(256)
void conv_silu_kernel(const float* __restrict__ xz, const float* __restrict__ w,
                      const float* __restrict__ cb, float* __restrict__ ucv)
{
  int i = blockIdx.x * blockDim.x + threadIdx.x;
  if (i >= BL * EXPD) return;
  int e = i % EXPD;
  int t = (i / EXPD) % SEQ;
  int b = i / (EXPD * SEQ);
  float acc = cb[e];
  #pragma unroll
  for (int k = 0; k < KCONV; ++k) {
    int tt = t - (KCONV - 1) + k;
    if (tt >= 0)
      acc += w[e * KCONV + k] * xz[(size_t)(b * SEQ + tt) * (2 * EXPD) + e];
  }
  ucv[i] = acc * (1.f / (1.f + __expf(-acc)));   // SiLU
}

// ---------------------------------------------------------------------------
// Selective scan + skip + gating. One thread per (b, e): h[16] in VGPRs,
// sequential over L=2048. Coalesced over e; B/C broadcast per (b,t).
// ---------------------------------------------------------------------------
__global__ __launch_bounds__(256)
void scan_kernel(const float* __restrict__ ucv, const float* __restrict__ dtb,
                 const float* __restrict__ dblb, const float* __restrict__ xz,
                 const float* __restrict__ A_log, const float* __restrict__ Dsk,
                 float* __restrict__ y)
{
  int ce = blockIdx.x * blockDim.x + threadIdx.x;   // b*E + e
  if (ce >= BATCH * EXPD) return;
  int b = ce / EXPD, e = ce % EXPD;
  float Av[NST], h[NST];
  #pragma unroll
  for (int n = 0; n < NST; ++n) {
    Av[n] = -__expf(A_log[(size_t)e * NST + n]);
    h[n]  = 0.f;
  }
  const float Dv = Dsk[e];
  for (int t = 0; t < SEQ; ++t) {
    size_t tok = (size_t)b * SEQ + t;
    float u  = ucv[tok * EXPD + e];
    float dt = dtb[tok * EXPD + e];
    const float4* bc = reinterpret_cast<const float4*>(dblb + tok * DBL_LD + RRANK);
    float Bv[NST], Cv[NST];
    #pragma unroll
    for (int q = 0; q < 4; ++q) {
      float4 v = bc[q];
      Bv[4*q+0] = v.x; Bv[4*q+1] = v.y; Bv[4*q+2] = v.z; Bv[4*q+3] = v.w;
    }
    #pragma unroll
    for (int q = 0; q < 4; ++q) {
      float4 v = bc[4 + q];
      Cv[4*q+0] = v.x; Cv[4*q+1] = v.y; Cv[4*q+2] = v.z; Cv[4*q+3] = v.w;
    }
    float du = dt * u;
    float yv = 0.f;
    #pragma unroll
    for (int n = 0; n < NST; ++n) {
      float dA = __expf(dt * Av[n]);      // v_exp_f32 (TRANS, co-executes)
      h[n] = dA * h[n] + du * Bv[n];
      yv  += h[n] * Cv[n];
    }
    float z  = xz[tok * (2 * EXPD) + EXPD + e];
    float sz = z * (1.f / (1.f + __expf(-z)));      // SiLU gate
    y[tok * EXPD + e] = (yv + u * Dv) * sz;
  }
}

// ---------------------------------------------------------------------------
extern "C" void kernel_launch(void* const* d_in, const int* in_sizes, int n_in,
                              void* d_out, int out_size, void* d_ws, size_t ws_size,
                              hipStream_t stream)
{
  const int*   ids    = (const int*)  d_in[0];
  const float* tok    = (const float*)d_in[1];
  const float* pos    = (const float*)d_in[2];
  const float* ln_g   = (const float*)d_in[3];
  const float* ln_b   = (const float*)d_in[4];
  const float* W_in   = (const float*)d_in[5];
  const float* conv_w = (const float*)d_in[6];
  const float* conv_b = (const float*)d_in[7];
  const float* W_x    = (const float*)d_in[8];
  const float* W_dt   = (const float*)d_in[9];
  const float* b_dt   = (const float*)d_in[10];
  const float* A_log  = (const float*)d_in[11];
  const float* D_skip = (const float*)d_in[12];
  const float* W_out  = (const float*)d_in[13];
  const float* lnf_g  = (const float*)d_in[14];
  const float* lnf_b  = (const float*)d_in[15];
  const float* W_head = (const float*)d_in[16];

  float* ws  = (float*)d_ws;
  float* x   = ws; ws += (size_t)BL * DIMM;      // residual stream
  float* xn  = ws; ws += (size_t)BL * DIMM;      // normalized
  float* xz  = ws; ws += (size_t)BL * 2 * EXPD;  // in-proj (u|z)
  float* ucv = ws; ws += (size_t)BL * EXPD;      // conv+silu(u)
  float* dbb = ws; ws += (size_t)BL * DBL_LD;    // [dt_lowrank | B | C]
  float* dtb = ws; ws += (size_t)BL * EXPD;      // softplus dt
  float* yb  = ws; ws += (size_t)BL * EXPD;      // scan output (gated)

  // bf16 transposed weights [N][K] (TDM source tiles)
  unsigned short* wsh   = (unsigned short*)ws;
  unsigned short* tWin  = wsh; wsh += (size_t)DEPTH * 2 * EXPD * DIMM;  // 3072x768
  unsigned short* tWx   = wsh; wsh += (size_t)DEPTH * DBL_LD * EXPD;    // 80x1536
  unsigned short* tWdt  = wsh; wsh += (size_t)DEPTH * EXPD * RRANK;     // 1536x48
  unsigned short* tWout = wsh; wsh += (size_t)DEPTH * DIMM * EXPD;      // 768x1536
  unsigned short* tWhd  = wsh; wsh += (size_t)VOC * DIMM;               // 1024x768

  dim3 blk(256);

  // ---- one-time (per launch) weight transpose + bf16 convert ----
  for (int l = 0; l < DEPTH; ++l) {
    transpose_bf16_kernel<<<(2*EXPD*DIMM + 255)/256, blk, 0, stream>>>(
        W_in  + (size_t)l * DIMM * 2 * EXPD, tWin  + (size_t)l * 2 * EXPD * DIMM, DIMM, 2*EXPD);
    transpose_bf16_kernel<<<(DBL_LD*EXPD + 255)/256, blk, 0, stream>>>(
        W_x   + (size_t)l * EXPD * DBL_LD,   tWx   + (size_t)l * DBL_LD * EXPD,   EXPD, DBL_LD);
    transpose_bf16_kernel<<<(EXPD*RRANK + 255)/256, blk, 0, stream>>>(
        W_dt  + (size_t)l * RRANK * EXPD,    tWdt  + (size_t)l * EXPD * RRANK,    RRANK, EXPD);
    transpose_bf16_kernel<<<(DIMM*EXPD + 255)/256, blk, 0, stream>>>(
        W_out + (size_t)l * EXPD * DIMM,     tWout + (size_t)l * DIMM * EXPD,     EXPD, DIMM);
  }
  transpose_bf16_kernel<<<(VOC*DIMM + 255)/256, blk, 0, stream>>>(
      W_head, tWhd, DIMM, VOC);

  embed_kernel<<<(BL * DIMM + 255) / 256, blk, 0, stream>>>(ids, tok, pos, x);

  for (int l = 0; l < DEPTH; ++l) {
    layernorm_kernel<<<BL, blk, 0, stream>>>(x, ln_g + l * DIMM, ln_b + l * DIMM, xn);

    // xz = xn @ W_in   (4096 x 3072 x 768)
    gemm_wmma_bf16<0, true><<<dim3((2*EXPD + 127)/128, (BL + 127)/128), blk, 0, stream>>>(
        xn, DIMM, tWin + (size_t)l * 2 * EXPD * DIMM,
        xz, 2 * EXPD, nullptr, BL, 2 * EXPD, DIMM);

    conv_silu_kernel<<<(BL * EXPD + 255) / 256, blk, 0, stream>>>(
        xz, conv_w + (size_t)l * EXPD * KCONV, conv_b + l * EXPD, ucv);

    // dbl = u @ W_x   (4096 x 80 x 1536)
    gemm_wmma_bf16<0, true><<<dim3((DBL_LD + 127)/128, (BL + 127)/128), blk, 0, stream>>>(
        ucv, EXPD, tWx + (size_t)l * DBL_LD * EXPD,
        dbb, DBL_LD, nullptr, BL, DBL_LD, EXPD);

    // dt = softplus(dbl[:, :48] @ W_dt + b_dt)   (4096 x 1536 x 48, K zero-padded)
    gemm_wmma_bf16<2, false><<<dim3((EXPD + 127)/128, (BL + 127)/128), blk, 0, stream>>>(
        dbb, DBL_LD, tWdt + (size_t)l * EXPD * RRANK,
        dtb, EXPD, b_dt + l * EXPD, BL, EXPD, RRANK);

    scan_kernel<<<(BATCH * EXPD + 255) / 256, blk, 0, stream>>>(
        ucv, dtb, dbb, xz, A_log + (size_t)l * EXPD * NST, D_skip + l * EXPD, yb);

    // x = x + y @ W_out   (4096 x 768 x 1536, fused residual)
    gemm_wmma_bf16<1, true><<<dim3((DIMM + 127)/128, (BL + 127)/128), blk, 0, stream>>>(
        yb, EXPD, tWout + (size_t)l * DIMM * EXPD,
        x, DIMM, x, BL, DIMM, EXPD);
  }

  layernorm_kernel<<<BL, blk, 0, stream>>>(x, lnf_g, lnf_b, xn);

  // logits = xn @ W_head   (4096 x 1024 x 768)
  gemm_wmma_bf16<0, true><<<dim3((VOC + 127)/128, (BL + 127)/128), blk, 0, stream>>>(
      xn, DIMM, tWhd, (float*)d_out, VOC, nullptr, BL, VOC, DIMM);

  (void)in_sizes; (void)n_in; (void)out_size; (void)ws_size;
}
